// PairNetLayer_symmetry_94489281040
// MI455X (gfx1250) — matrix operand order repaired and checked by reading, not compiled, not run on hardware
//
#include <hip/hip_runtime.h>
#include <math.h>

#define NNODE 512
#define M0 128
#define M1 64
#define MT0 192
#define WN 256
#define EA 32
#define INVD 8

typedef __attribute__((ext_vector_type(16))) __bf16 v16bf;
typedef __attribute__((ext_vector_type(8)))  __bf16 v8bf;
typedef __attribute__((ext_vector_type(8)))  float  v8f;

// ---- constants --------------------------------------------------------------
#define INV_SQRT128 0.08838834764831845f
#define INV_SQRT192 0.07216878364870323f
#define INV_SQRT64  0.125f
#define INV_SQRT32  0.17677669529663687f
#define INV_SQRT8   0.35355339059327373f
#define INV_SQRT3   0.5773502691896258f
#define INV_SQRT2   0.7071067811865476f
#define LN2F        0.6931471805599453f

// frag table bases (in 512-bf16 tiles): fcs_W1(6x8), fcs_W2(4x16), ng_W1(8x16),
// ng_W2(8x16), res_W0(6x8), res_W1(2x4)
#define F1 0
#define F2 48
#define F3 112
#define F4 240
#define F5 368
#define F6 416
#define NFRAG 424

#define LDA 392   // bf16 row stride, A ping buffer (max 384 cols)
#define LDB 264   // bf16 row stride, A pong buffer (max 256 cols)

// silu via v_rcp_f32 (avoids IEEE division macro expansion)
__device__ __forceinline__ float siluf(float x){
  return x * __builtin_amdgcn_rcpf(1.f + __expf(-x));
}

__device__ __forceinline__ v8f wmma_bf16(v16bf a, v16bf b, v8f c){
  return __builtin_amdgcn_wmma_f32_16x16x32_bf16(false, a, false, b, (short)0, c, false, false);
}

// Build 16-bit A fragment, contiguous k order (same as k_frag B):
// lanes 0-15: row=lane, K = kbase+0..15
// lanes16-31: row=lane-16, K = kbase+16..31
__device__ __forceinline__ v16bf load_a(const __bf16* As, int lda, int lane, int kbase){
  int r  = lane & 15;
  int kg = (lane >> 4) * 16;
  const __bf16* p = As + r * lda + kbase + kg;
  v8bf lo = *(const v8bf*)p;
  v8bf hi = *(const v8bf*)(p + 8);
  v16bf a;
#pragma unroll
  for (int j = 0; j < 8; ++j){ a[j] = lo[j]; a[j+8] = hi[j]; }
  return a;
}

// ---- node prep: na0 = e3_linear(node_attr, W_inner0, b_inner0, W_inner1) ----
__global__ void k_na0(const float* __restrict__ na, const float* __restrict__ W0,
                      const float* __restrict__ b0, const float* __restrict__ W1,
                      float* __restrict__ out){
  int n = blockIdx.x, t = threadIdx.x;           // 320 threads
  const float* x = na + (size_t)n * 320;
  float acc;
  if (t < M0){
    float s = 0.f;
#pragma unroll 4
    for (int k = 0; k < M0; ++k) s += x[k] * W0[k * M0 + t];
    acc = s * INV_SQRT128 + b0[t];
  } else {
    int c = t - M0, v = c / 3, i = c % 3;
    float s = 0.f;
#pragma unroll 4
    for (int u = 0; u < M1; ++u) s += x[M0 + 3*u + i] * W1[u * M1 + v];
    acc = s * INV_SQRT64;
  }
  out[(size_t)n * 320 + t] = acc;
}

// ---- node prep: h = e3_linear(norm_gate(node_attr, gp_*), W_in*) ------------
__global__ void k_h(const float* __restrict__ na,
                    const float* __restrict__ gW1, const float* __restrict__ gb1,
                    const float* __restrict__ gW2, const float* __restrict__ gb2,
                    const float* __restrict__ W0,  const float* __restrict__ b0,
                    const float* __restrict__ W1,  float* __restrict__ out){
  __shared__ float xb[MT0], g1[MT0], g2[MT0], y[320];
  int n = blockIdx.x, t = threadIdx.x;           // 320 threads
  const float* x = na + (size_t)n * 320;
  if (t < M0) xb[t] = x[t];
  else if (t < MT0){
    int u = t - M0;
    float a = x[M0+3*u], b = x[M0+3*u+1], c = x[M0+3*u+2];
    xb[t] = sqrtf(a*a + b*b + c*c + 1e-12f);
  }
  __syncthreads();
  if (t < MT0){
    float s = gb1[t];
#pragma unroll 4
    for (int k = 0; k < MT0; ++k) s += xb[k] * gW1[k * MT0 + t];
    g1[t] = siluf(s);
  }
  __syncthreads();
  if (t < MT0){
    float s = gb2[t];
#pragma unroll 4
    for (int k = 0; k < MT0; ++k) s += g1[k] * gW2[k * MT0 + t];
    g2[t] = s;
  }
  __syncthreads();
  if (t < M0) y[t] = g2[t];
  else { int u = (t - M0) / 3; y[t] = x[t] * g2[M0 + u]; }
  __syncthreads();
  float acc;
  if (t < M0){
    float s = 0.f;
#pragma unroll 4
    for (int k = 0; k < M0; ++k) s += y[k] * W0[k * M0 + t];
    acc = s * INV_SQRT128 + b0[t];
  } else {
    int c = t - M0, v = c / 3, i = c % 3;
    float s = 0.f;
#pragma unroll 4
    for (int u = 0; u < M1; ++u) s += y[M0 + 3*u + i] * W1[u * M1 + v];
    acc = s * INV_SQRT64;
  }
  out[(size_t)n * 320 + t] = acc;
}

// ---- weight prep: bf16 B fragments, contiguous k order (k = 16*(lane/16) + j) -----------
__global__ void k_frag(const float* __restrict__ W, int N, float scale,
                       __bf16* __restrict__ out){
  int tile = blockIdx.x, lane = threadIdx.x;     // 32 threads
  int NT = N >> 4;
  int ko = tile / NT, no = tile % NT;
  int col = no * 16 + (lane & 15);
  int kb  = ko * 32 + ((lane >> 4) << 4);
  __bf16* o = out + ((size_t)tile * 32 + lane) * 16;
#pragma unroll
  for (int j = 0; j < 16; ++j) o[j] = (__bf16)(W[(size_t)(kb + j) * N + col] * scale);
}

// ---- fused edge kernel ------------------------------------------------------
__global__ __launch_bounds__(256, 2)
void k_edge(const float* __restrict__ na0, const float* __restrict__ h,
            const v16bf* __restrict__ frags,
            const float* __restrict__ edge_attr, const float* __restrict__ npa,
            const float* __restrict__ fcnp_W1, const float* __restrict__ fcnp_W2,
            const float* __restrict__ fcs_b1, const float* __restrict__ fcs_b2,
            const float* __restrict__ ng_b1,  const float* __restrict__ ng_b2,
            const float* __restrict__ res_b0,
            const int* __restrict__ dst, const int* __restrict__ src,
            float* __restrict__ out){
  __shared__ __attribute__((aligned(16))) __bf16 AsA[16 * LDA];
  __shared__ __attribute__((aligned(16))) __bf16 AsB[16 * LDB];
  __shared__ float vbuf[16 * 192];
  __shared__ float w2s[8 * 256];
  __shared__ float sspb[16 * 8];
  __shared__ int   sIdx[16], dIdx[16];

  const int t = threadIdx.x;
  const int lane = t & 31, wv = t >> 5;
  const int e0 = blockIdx.x * 16;
  const int ncl = lane & 15;          // local col within a 16x16 tile
  const int rb  = (lane >> 4) * 8;    // D row base for this lane group

  // stage -1: edge indices into LDS
  if (t < 16)      sIdx[t]      = src[e0 + t];
  else if (t < 32) dIdx[t - 16] = dst[e0 + t - 16];
  __syncthreads();

  // stage 0a: ssp = softplus(edge_attr @ fcnp_W1/sqrt(32)) - ln2   (16x8)
  if (t < 128){
    int e = t >> 3, j = t & 7;
    const float* ea = edge_attr + (size_t)(e0 + e) * EA;
    float s = 0.f;
#pragma unroll
    for (int k = 0; k < EA; ++k) s += ea[k] * fcnp_W1[k * INVD + j];
    s *= INV_SQRT32;
    float sp = (s > 20.f) ? s : __logf(1.f + __expf(s));
    sspb[t] = sp - LN2F;
  }
  // stage 0b: stage fcnp_W2/sqrt(8) into LDS (8x256)
  for (int q = t; q < 8 * 256; q += 256) w2s[q] = fcnp_W2[q] * INV_SQRT8;

  // stage 0c: A1 = s0 (16x192) bf16 from na0 gathers
  for (int p = t; p < 16 * MT0; p += 256){
    int e = p / MT0, c = p % MT0;
    const float* xs = na0 + (size_t)sIdx[e] * 320;
    const float* xd = na0 + (size_t)dIdx[e] * 320;
    float val;
    if (c < M0) val = 0.5f * (xd[c] + xs[c]);
    else {
      int u = c - M0;
      val = (xd[M0+3*u]*xs[M0+3*u] + xd[M0+3*u+1]*xs[M0+3*u+1] +
             xd[M0+3*u+2]*xs[M0+3*u+2]) * INV_SQRT3;
    }
    AsA[e * LDA + c] = (__bf16)val;
  }
  __syncthreads();

  // M1: s0(16x192) @ fcs_W1 -> +b -> silu -> A2(16x128) in AsB
  {
    v8f acc = {};
#pragma unroll
    for (int ko = 0; ko < 6; ++ko){
      v16bf a = load_a(AsA, LDA, lane, ko * 32);
      v16bf b = frags[(F1 + ko * 8 + wv) * 32 + lane];
      acc = wmma_bf16(a, b, acc);
    }
    int ncol = wv * 16 + ncl;
    float bias = fcs_b1[ncol];
#pragma unroll
    for (int r = 0; r < 8; ++r)
      AsB[(r + rb) * LDB + ncol] = (__bf16)siluf(acc[r] + bias);
  }
  __syncthreads();

  // M2: A2 @ fcs_W2 -> w_s0; w = w_edge*w_s0; pair elementwise -> A3(AsA)+vbuf
#pragma unroll
  for (int tt = 0; tt < 2; ++tt){
    int no = wv + tt * 8;
    v8f acc = {};
#pragma unroll
    for (int ko = 0; ko < 4; ++ko){
      v16bf a = load_a(AsB, LDB, lane, ko * 32);
      v16bf b = frags[(F2 + ko * 16 + no) * 32 + lane];
      acc = wmma_bf16(a, b, acc);
    }
    int ncol = no * 16 + ncl;
    float b2 = fcs_b2[ncol];
#pragma unroll
    for (int r = 0; r < 8; ++r){
      int e = r + rb;
      float ws0 = acc[r] + b2;
      float we = 0.f;
#pragma unroll
      for (int j = 0; j < 8; ++j) we += sspb[e * 8 + j] * w2s[j * 256 + ncol];
      float wval = we * ws0;
      const float* hs = h + (size_t)sIdx[e] * 320;
      const float* hd = h + (size_t)dIdx[e] * 320;
      if (ncol < M0){
        AsA[e * LDA + ncol] = (__bf16)(wval * hs[ncol] * hd[ncol]);
      } else if (ncol < MT0){
        int u = ncol - M0;
        float d3 = hs[M0+3*u]*hd[M0+3*u] + hs[M0+3*u+1]*hd[M0+3*u+1] +
                   hs[M0+3*u+2]*hd[M0+3*u+2];
        AsA[e * LDA + ncol] = (__bf16)(wval * d3 * INV_SQRT3);
      } else {
        int u = ncol - MT0;
        float ax = hs[M0+3*u], ay = hs[M0+3*u+1], az = hs[M0+3*u+2];
        float bx = hd[M0+3*u], by = hd[M0+3*u+1], bz = hd[M0+3*u+2];
        float k = wval * INV_SQRT2;
        float vx = k * (ay*bz - az*by);
        float vy = k * (az*bx - ax*bz);
        float vz = k * (ax*by - ay*bx);
        vbuf[e * 192 + 3*u    ] = vx;
        vbuf[e * 192 + 3*u + 1] = vy;
        vbuf[e * 192 + 3*u + 2] = vz;
        AsA[e * LDA + ncol] = (__bf16)sqrtf(vx*vx + vy*vy + vz*vz + 1e-12f);
      }
    }
  }
  __syncthreads();

  // M3: A3(16x256) @ ng_W1 -> +b -> silu -> A4(16x256) in AsB
#pragma unroll
  for (int tt = 0; tt < 2; ++tt){
    int no = wv + tt * 8;
    v8f acc = {};
#pragma unroll
    for (int ko = 0; ko < 8; ++ko){
      v16bf a = load_a(AsA, LDA, lane, ko * 32);
      v16bf b = frags[(F3 + ko * 16 + no) * 32 + lane];
      acc = wmma_bf16(a, b, acc);
    }
    int ncol = no * 16 + ncl;
    float bias = ng_b1[ncol];
#pragma unroll
    for (int r = 0; r < 8; ++r)
      AsB[(r + rb) * LDB + ncol] = (__bf16)siluf(acc[r] + bias);
  }
  __syncthreads();

  // M4: A4 @ ng_W2 -> g; gate -> A5 = [g_s(192) | v*g_v packed per-axis(3x64)]
#pragma unroll
  for (int tt = 0; tt < 2; ++tt){
    int no = wv + tt * 8;
    v8f acc = {};
#pragma unroll
    for (int ko = 0; ko < 8; ++ko){
      v16bf a = load_a(AsB, LDB, lane, ko * 32);
      v16bf b = frags[(F4 + ko * 16 + no) * 32 + lane];
      acc = wmma_bf16(a, b, acc);
    }
    int ncol = no * 16 + ncl;
    float bias = ng_b2[ncol];
#pragma unroll
    for (int r = 0; r < 8; ++r){
      int e = r + rb;
      float g = acc[r] + bias;
      if (ncol < MT0){
        AsA[e * LDA + ncol] = (__bf16)g;
      } else {
        int u = ncol - MT0;
#pragma unroll
        for (int i = 0; i < 3; ++i)
          AsA[e * LDA + MT0 + 64 * i + u] = (__bf16)(vbuf[e * 192 + 3*u + i] * g);
      }
    }
  }
  __syncthreads();

  // M5 scalar: A5[:, :192] @ res_W0/sqrt(192) + res_b0 -> out cols 0..127
  {
    v8f acc = {};
#pragma unroll
    for (int ko = 0; ko < 6; ++ko){
      v16bf a = load_a(AsA, LDA, lane, ko * 32);
      v16bf b = frags[(F5 + ko * 8 + wv) * 32 + lane];
      acc = wmma_bf16(a, b, acc);
    }
    int ncol = wv * 16 + ncl;
    float bias = res_b0[ncol];
#pragma unroll
    for (int r = 0; r < 8; ++r){
      size_t o = (size_t)(e0 + r + rb) * 320 + ncol;
      out[o] = acc[r] + bias + npa[o];
    }
  }
  // M5 vector: per axis i, A5[:, 192+64i:+64] @ res_W1/8 -> out col 128+3v+i
  for (int tv = wv; tv < 12; tv += 8){
    int i = tv >> 2, no = tv & 3;
    v8f acc = {};
#pragma unroll
    for (int ko = 0; ko < 2; ++ko){
      v16bf a = load_a(AsA, LDA, lane, MT0 + 64 * i + ko * 32);
      v16bf b = frags[(F6 + ko * 4 + no) * 32 + lane];
      acc = wmma_bf16(a, b, acc);
    }
    int vcol = no * 16 + ncl;
#pragma unroll
    for (int r = 0; r < 8; ++r){
      size_t o = (size_t)(e0 + r + rb) * 320 + (M0 + 3 * vcol + i);
      out[o] = acc[r] + npa[o];
    }
  }
}

// ---- launcher ---------------------------------------------------------------
extern "C" void kernel_launch(void* const* d_in, const int* in_sizes, int n_in,
                              void* d_out, int out_size, void* d_ws, size_t ws_size,
                              hipStream_t stream){
  const float* node_attr = (const float*)d_in[0];
  const float* edge_attr = (const float*)d_in[1];
  const float* npa       = (const float*)d_in[2];
  const float* W_inner0  = (const float*)d_in[3];
  const float* b_inner0  = (const float*)d_in[4];
  const float* W_inner1  = (const float*)d_in[5];
  const float* W_in0     = (const float*)d_in[6];
  const float* b_in0     = (const float*)d_in[7];
  const float* W_in1     = (const float*)d_in[8];
  const float* gp_W1     = (const float*)d_in[9];
  const float* gp_b1     = (const float*)d_in[10];
  const float* gp_W2     = (const float*)d_in[11];
  const float* gp_b2     = (const float*)d_in[12];
  const float* fcnp_W1   = (const float*)d_in[13];
  const float* fcnp_W2   = (const float*)d_in[14];
  const float* fcs_W1    = (const float*)d_in[15];
  const float* fcs_b1    = (const float*)d_in[16];
  const float* fcs_W2    = (const float*)d_in[17];
  const float* fcs_b2    = (const float*)d_in[18];
  const float* ng_W1     = (const float*)d_in[19];
  const float* ng_b1     = (const float*)d_in[20];
  const float* ng_W2     = (const float*)d_in[21];
  const float* ng_b2     = (const float*)d_in[22];
  const float* res_W0    = (const float*)d_in[23];
  const float* res_b0    = (const float*)d_in[24];
  const float* res_W1    = (const float*)d_in[25];
  const int*   dst       = (const int*)d_in[26];
  const int*   src       = (const int*)d_in[27];
  float* out = (float*)d_out;
  const int E = in_sizes[26];

  // workspace layout
  float*  na0   = (float*)d_ws;                                   // 512*320 f32
  float*  hbuf  = (float*)((char*)d_ws + 512 * 320 * 4);          // 512*320 f32
  __bf16* fr    = (__bf16*)((char*)d_ws + 2 * 512 * 320 * 4);     // 424*512 bf16

  k_na0<<<NNODE, 320, 0, stream>>>(node_attr, W_inner0, b_inner0, W_inner1, na0);
  k_h  <<<NNODE, 320, 0, stream>>>(node_attr, gp_W1, gp_b1, gp_W2, gp_b2,
                                   W_in0, b_in0, W_in1, hbuf);

  k_frag<<< 48, 32, 0, stream>>>(fcs_W1, 128, 1.f,         fr + (size_t)F1 * 512);
  k_frag<<< 64, 32, 0, stream>>>(fcs_W2, 256, 1.f,         fr + (size_t)F2 * 512);
  k_frag<<<128, 32, 0, stream>>>(ng_W1,  256, 1.f,         fr + (size_t)F3 * 512);
  k_frag<<<128, 32, 0, stream>>>(ng_W2,  256, 1.f,         fr + (size_t)F4 * 512);
  k_frag<<< 48, 32, 0, stream>>>(res_W0, 128, INV_SQRT192, fr + (size_t)F5 * 512);
  k_frag<<<  8, 32, 0, stream>>>(res_W1,  64, INV_SQRT64,  fr + (size_t)F6 * 512);

  k_edge<<<E / 16, 256, 0, stream>>>(na0, hbuf, (const v16bf*)fr,
                                     edge_attr, npa, fcnp_W1, fcnp_W2,
                                     fcs_b1, fcs_b2, ng_b1, ng_b2, res_b0,
                                     dst, src, out);
}
